// NetworkGRU_reimplement_28192165331635
// MI455X (gfx1250) — matrix operand order repaired and checked by reading, not compiled, not run on hardware
//
#include <hip/hip_runtime.h>
#include <hip/hip_bf16.h>

// ---------------------------------------------------------------------------
// GRU on gfx1250 (MI455X): persistent 16-workgroup scan kernel.
//   - W_h / W_i slices resident in LDS for the whole scan (144 KB)
//   - h staged into LDS each step via GLOBAL_LOAD_ASYNC_TO_LDS_B128
//     (ASYNCcnt), overlapped with the input-projection WMMAs
//   - bf16 WMMA 16x16x32, fp32 accumulate
//   - double-buffered h (fp32 + bf16 mirror) in workspace, L2-resident
//   - device atomic grid barrier + cluster split-barrier (NOP unclustered)
// ---------------------------------------------------------------------------

#define SEQLEN   512
#define NBATCH   64
#define NEMB     256
#define NHID     512
#define NWG      16          // workgroups; each owns 32 columns of H
#define WGCOLS   32
#define NTHREADS 256         // 8 wave32s

#if defined(__has_builtin)
#if __has_builtin(__builtin_amdgcn_global_load_async_to_lds_b128) && \
    __has_builtin(__builtin_amdgcn_s_wait_asynccnt)
#define USE_ASYNC_LDS 1
#endif
#endif
#ifndef USE_ASYNC_LDS
#define USE_ASYNC_LDS 0
#endif

typedef __attribute__((ext_vector_type(16))) __bf16        v16bf;
typedef __attribute__((ext_vector_type(8)))  float         v8f;
typedef __attribute__((ext_vector_type(4)))  unsigned int  v4u;
typedef __attribute__((ext_vector_type(4)))  int           v4i;

union Frag {
  v16bf bf;
  v4u   q[2];
};

__device__ __forceinline__ unsigned short f2bf(float f) {
  unsigned int u = __float_as_uint(f);
  u += 0x7fffu + ((u >> 16) & 1u);         // round to nearest even
  return (unsigned short)(u >> 16);
}

__device__ __forceinline__ float sigmf(float x) {
  return 1.0f / (1.0f + __expf(-x));
}
__device__ __forceinline__ float tanh_fast(float x) {
  return 2.0f / (1.0f + __expf(-2.0f * x)) - 1.0f;
}

#if USE_ASYNC_LDS
// one 16-byte async DMA transfer: global -> LDS, tracked on ASYNCcnt
__device__ __forceinline__ void async_b128(const unsigned short* gsrc,
                                           unsigned short* ldst) {
  __builtin_amdgcn_global_load_async_to_lds_b128(
      (__attribute__((address_space(1))) v4i*)(gsrc),
      (__attribute__((address_space(3))) v4i*)(ldst),
      0, 0);
}
#endif

// ---------------------------------------------------------------------------
__global__ void gru_prep(float* h32, unsigned short* h16, unsigned int* cnt) {
  int i = blockIdx.x * blockDim.x + threadIdx.x;
  if (i < 2 * NBATCH * NHID) {
    h32[i] = 0.0f;
    h16[i] = 0;
  }
  if (i == 0) *cnt = 0u;
}

// ---------------------------------------------------------------------------
__global__ __launch_bounds__(NTHREADS) void gru_scan(
    const int*   __restrict__ words, const float* __restrict__ emb,
    const float* __restrict__ Wir,   const float* __restrict__ bir,
    const float* __restrict__ Wiz,   const float* __restrict__ biz,
    const float* __restrict__ Win,   const float* __restrict__ bin_,
    const float* __restrict__ Whr,   const float* __restrict__ bhr,
    const float* __restrict__ Whz,   const float* __restrict__ bhz,
    const float* __restrict__ Whn,   const float* __restrict__ bhn,
    const float* __restrict__ Wfc,   const float* __restrict__ bfc,
    float* __restrict__ h32, unsigned short* __restrict__ h16,
    unsigned int* __restrict__ cnt, float* __restrict__ out)
{
  extern __shared__ unsigned short smem[];
  unsigned short* whS = smem;                              // [3][32][512] bf16
  unsigned short* wiS = whS + 3 * WGCOLS * NHID;           // [3][32][256] bf16
  unsigned short* xS  = wiS + 3 * WGCOLS * NEMB;           // [64][256]    bf16
#if USE_ASYNC_LDS
  unsigned short* hS  = xS + NBATCH * NEMB;                // [64][512]    bf16
#endif

  const int tid  = threadIdx.x;
  const int wg   = blockIdx.x;          // 0..15, owns cols [wg*32, wg*32+32)
  const int lane = tid & 31;
  const int wv   = tid >> 5;            // wave id 0..7
  const int mt   = wv >> 1;             // 4 M-tiles of 16 batch rows
  const int nt   = wv & 1;              // 2 N-tiles of 16 columns

  // ---- stage weight slices into LDS (fp32 -> bf16), resident for the scan
  for (int i = tid; i < 3 * WGCOLS * NHID; i += NTHREADS) {
    int g = i / (WGCOLS * NHID);
    int r = (i / NHID) % WGCOLS;
    int k = i % NHID;
    const float* src = (g == 0) ? Whr : ((g == 1) ? Whz : Whn);
    whS[i] = f2bf(src[(wg * WGCOLS + r) * NHID + k]);
  }
  for (int i = tid; i < 3 * WGCOLS * NEMB; i += NTHREADS) {
    int g = i / (WGCOLS * NEMB);
    int r = (i / NEMB) % WGCOLS;
    int k = i % NEMB;
    const float* src = (g == 0) ? Wir : ((g == 1) ? Wiz : Win);
    wiS[i] = f2bf(src[(wg * WGCOLS + r) * NEMB + k]);
  }
  __syncthreads();

  // ---- per-wave fragment geometry (CDNA5 WMMA VGPR layouts)
  const int m0     = mt * 16;
  const int aRow   = m0 + (lane & 15);              // A: row per lane
  const int aKoff  = (lane & 16) ? 16 : 0;          // A: K half select
  const int bKoff  = (lane & 16) ? 16 : 0;          // B: K half select
  const int nSlice = nt * 16 + (lane & 15);         // B: weight row in slice
  const int nAbs   = wg * WGCOLS + nSlice;          // absolute H column

  const float c_bir = bir[nAbs], c_biz = biz[nAbs], c_bin = bin_[nAbs];
  const float c_bhr = bhr[nAbs], c_bhz = bhz[nAbs], c_bhn = bhn[nAbs];

  // ---- sequential scan
  for (int s = 0; s < SEQLEN; ++s) {
    const int rb = s & 1;
    const unsigned short* hRd   = h16 + rb * NBATCH * NHID;
    const float*          h32Rd = h32 + rb * NBATCH * NHID;
    float*                h32Wr = h32 + (rb ^ 1) * NBATCH * NHID;
    unsigned short*       h16Wr = h16 + (rb ^ 1) * NBATCH * NHID;

#if USE_ASYNC_LDS
    // async DMA of h (bf16, 64KB) from L2 into LDS; overlapped with the
    // x-gather and input-projection WMMAs below. Tracked on ASYNCcnt.
    for (int c = tid; c < (NBATCH * NHID) / 8; c += NTHREADS) {
      async_b128(hRd + c * 8, hS + c * 8);
    }
#endif

    // gather x_s = emb[words[:, s]] into LDS as bf16 (redundant per WG; L2 hit)
    {
      int b  = tid >> 2;
      int qq = tid & 3;
      int word = words[b * SEQLEN + s];
      const float* src = emb + (size_t)word * NEMB + qq * 64;
      unsigned short* dst = xS + b * NEMB + qq * 64;
#pragma unroll
      for (int j = 0; j < 64; j += 4) {
        float4 f = *(const float4*)(src + j);
        dst[j + 0] = f2bf(f.x); dst[j + 1] = f2bf(f.y);
        dst[j + 2] = f2bf(f.z); dst[j + 3] = f2bf(f.w);
      }
      if (qq == 0 && s + 1 < SEQLEN) {   // warm L2/WGP$ for the next step
        int wn = words[b * SEQLEN + s + 1];
        __builtin_prefetch(emb + (size_t)wn * NEMB, 0, 1);
      }
    }
    __syncthreads();

    v8f ar = {}, az = {}, an = {};
    // input projections: x_s[16x256] x W_i^T[256x16], K chunks of 32
#pragma unroll 2
    for (int kc = 0; kc < NEMB; kc += 32) {
      Frag a;
      const unsigned short* ap = xS + aRow * NEMB + kc + aKoff;
      a.q[0] = *(const v4u*)(ap);
      a.q[1] = *(const v4u*)(ap + 8);
      const unsigned short* bp = wiS + nSlice * NEMB + kc + bKoff;
      Frag b0, b1, b2;
      b0.q[0] = *(const v4u*)(bp);
      b0.q[1] = *(const v4u*)(bp + 8);
      b1.q[0] = *(const v4u*)(bp + WGCOLS * NEMB);
      b1.q[1] = *(const v4u*)(bp + WGCOLS * NEMB + 8);
      b2.q[0] = *(const v4u*)(bp + 2 * WGCOLS * NEMB);
      b2.q[1] = *(const v4u*)(bp + 2 * WGCOLS * NEMB + 8);
      ar = __builtin_amdgcn_wmma_f32_16x16x32_bf16(false, a.bf, false, b0.bf,
                                                   (short)0, ar, false, false);
      az = __builtin_amdgcn_wmma_f32_16x16x32_bf16(false, a.bf, false, b1.bf,
                                                   (short)0, az, false, false);
      an = __builtin_amdgcn_wmma_f32_16x16x32_bf16(false, a.bf, false, b2.bf,
                                                   (short)0, an, false, false);
    }

#if USE_ASYNC_LDS
    // h tile must be fully landed in LDS before recurrent A-fragment reads
    __builtin_amdgcn_s_wait_asynccnt(0);
    __syncthreads();
#endif

    v8f hr = {}, hz = {}, hn = {};
    // recurrent: h[16x512] x W_h^T[512x16]
#pragma unroll 2
    for (int kc = 0; kc < NHID; kc += 32) {
      Frag a;
#if USE_ASYNC_LDS
      const unsigned short* ap = hS + aRow * NHID + kc + aKoff;
#else
      const unsigned short* ap = hRd + aRow * NHID + kc + aKoff;
#endif
      a.q[0] = *(const v4u*)(ap);
      a.q[1] = *(const v4u*)(ap + 8);
      const unsigned short* bp = whS + nSlice * NHID + kc + bKoff;
      Frag b0, b1, b2;
      b0.q[0] = *(const v4u*)(bp);
      b0.q[1] = *(const v4u*)(bp + 8);
      b1.q[0] = *(const v4u*)(bp + WGCOLS * NHID);
      b1.q[1] = *(const v4u*)(bp + WGCOLS * NHID + 8);
      b2.q[0] = *(const v4u*)(bp + 2 * WGCOLS * NHID);
      b2.q[1] = *(const v4u*)(bp + 2 * WGCOLS * NHID + 8);
      hr = __builtin_amdgcn_wmma_f32_16x16x32_bf16(false, a.bf, false, b0.bf,
                                                   (short)0, hr, false, false);
      hz = __builtin_amdgcn_wmma_f32_16x16x32_bf16(false, a.bf, false, b1.bf,
                                                   (short)0, hz, false, false);
      hn = __builtin_amdgcn_wmma_f32_16x16x32_bf16(false, a.bf, false, b2.bf,
                                                   (short)0, hn, false, false);
    }

    // elementwise GRU cell on the 16x16 C tile; C layout: M=v+8*laneHi, N=lane&15
    const int mBase = m0 + ((lane & 16) ? 8 : 0);
#pragma unroll
    for (int v = 0; v < 8; ++v) {
      int mrow = mBase + v;
      float rt   = sigmf(ar[v] + c_bir + hr[v] + c_bhr);
      float zt   = sigmf(az[v] + c_biz + hz[v] + c_bhz);
      float ntv  = tanh_fast(an[v] + c_bin + (hn[v] + c_bhn) * rt);
      float hOld = h32Rd[mrow * NHID + nAbs];
      float hNew = (1.0f - zt) * ntv + zt * hOld;
      h32Wr[mrow * NHID + nAbs] = hNew;
      h16Wr[mrow * NHID + nAbs] = f2bf(hNew);
    }

    // ---- grid-wide step barrier: atomic counter (always works) +
    //      cluster split-barrier (HW path if clustered, NOP otherwise)
    __syncthreads();
    if (tid == 0) {
      __threadfence();
      atomicAdd(cnt, 1u);
      unsigned int tgt = (unsigned int)(NWG * (s + 1));
      while (__hip_atomic_load(cnt, __ATOMIC_ACQUIRE,
                               __HIP_MEMORY_SCOPE_AGENT) < tgt) {
        __builtin_amdgcn_s_sleep(1);
      }
    }
    if (wv == 0) __builtin_amdgcn_s_cluster_barrier();
    __syncthreads();
  }

  // ---- final FC + sigmoid, and emit hidden; after 512 steps h lives in buf 0
  if (wg == 0) {
    const float* hFin = h32;   // (511&1)^1 == 0
    if (tid < NBATCH) {
      float acc = 0.0f;
      for (int k = 0; k < NHID; ++k) acc += hFin[tid * NHID + k] * Wfc[k];
      out[tid] = sigmf(acc + bfc[0]);
    }
    for (int i = tid; i < NBATCH * NHID; i += NTHREADS)
      out[NBATCH + i] = hFin[i];
  }
}

// ---------------------------------------------------------------------------
extern "C" void kernel_launch(void* const* d_in, const int* in_sizes, int n_in,
                              void* d_out, int out_size, void* d_ws,
                              size_t ws_size, hipStream_t stream) {
  const int*   words = (const int*)  d_in[0];
  const float* emb   = (const float*)d_in[1];
  const float* Wir   = (const float*)d_in[2];
  const float* bir   = (const float*)d_in[3];
  const float* Wiz   = (const float*)d_in[4];
  const float* biz   = (const float*)d_in[5];
  const float* Win   = (const float*)d_in[6];
  const float* bin_  = (const float*)d_in[7];
  const float* Whr   = (const float*)d_in[8];
  const float* bhr   = (const float*)d_in[9];
  const float* Whz   = (const float*)d_in[10];
  const float* bhz   = (const float*)d_in[11];
  const float* Whn   = (const float*)d_in[12];
  const float* bhn   = (const float*)d_in[13];
  const float* Wfc   = (const float*)d_in[14];
  const float* bfc   = (const float*)d_in[15];
  float* out = (float*)d_out;

  // workspace: double-buffered h (fp32 + bf16) + barrier counter (~384 KB)
  float*          h32 = (float*)d_ws;
  unsigned short* h16 = (unsigned short*)(h32 + 2 * NBATCH * NHID);
  unsigned int*   cnt = (unsigned int*)(h16 + 2 * NBATCH * NHID);

  gru_prep<<<(2 * NBATCH * NHID + 255) / 256, 256, 0, stream>>>(h32, h16, cnt);

  size_t ldsShorts = (size_t)(3 * WGCOLS * NHID + 3 * WGCOLS * NEMB +
                              NBATCH * NEMB);
#if USE_ASYNC_LDS
  ldsShorts += (size_t)NBATCH * NHID;    // h staging tile
#endif
  size_t ldsBytes = ldsShorts * sizeof(unsigned short);   // 176 KB / 240 KB
  gru_scan<<<NWG, NTHREADS, ldsBytes, stream>>>(
      words, emb, Wir, bir, Wiz, biz, Win, bin_,
      Whr, bhr, Whz, bhz, Whn, bhn, Wfc, bfc, h32, h16, cnt, out);
}
